// AngleGNNLayer_21938692948604
// MI455X (gfx1250) — matrix operand order repaired and checked by reading, not compiled, not run on hardware
//
#include <hip/hip_runtime.h>

// ============================================================================
// AngleGNN layer for MI455X (gfx1250, wave32, WMMA).
//
// Edge branch rewritten as ONE implicit GEMM:
//   msg(E,32) = Z(E,1056) @ W(1056,32),  Z[e, kh*32+i] = h[e,kh] * x[col[e], i]
// where W rows 0..1023 = eW2 (row-major (1024,32) view) and rows 1024..1055 =
// eb2 reshaped (32,32) (bias folded as an extra K-block with h == 1 -> its A
// operand is the cached x fragment itself, no scaling needed).
// Z is never materialized: each 16x32 A-chunk is rebuilt in registers from a
// cached x-fragment scaled by h[e,kh] (v_pk_mul_f16 w/ op_sel broadcast), then
// fed to v_wmma_f32_16x16x32_f16 with fp32 accumulation.
//
// K-blocks are processed in pairs with two independent A-fragment register
// sets so the 4-coexec-slot WMMA->VALU WAR hazard on the A registers
// (CDNA5 ISA §7.12.1) is covered by useful work, not v_nops.
//
// B-fragments are pre-swizzled once into the ISA B-matrix lane layout
// (B 32x16 f16 -> lane = g*16+n, slot t=(v,p) holds K = (v>=4 ? 16 : 0) + 2*(v&3) + p + 8g) and
// re-read by all waves as clause'd global_load_b128 pairs from WGP$/L2.
// ============================================================================

typedef __attribute__((ext_vector_type(16))) _Float16 v16h;
typedef __attribute__((ext_vector_type(8)))  float    v8f;

#define CH 32  // IN_CH == OUT_CH == HIDDEN == 32

#define WMMA_F16(A, B, C)                                                     \
  __builtin_amdgcn_wmma_f32_16x16x32_f16(false, (A), false, (B), (short)0,    \
                                         (C), false, false)

// A-matrix (16x32, 16-bit) lane layout, CDNA5 ISA §7.12.2:
// lane = g*16 + m ; slot t = 2v+p holds K = (v>=4 ? 16 : 0) + 2*(v&3) + p + 8g
__device__ __forceinline__ int a_kidx(int t, int g) {
  int v = t >> 1, p = t & 1;
  return ((v & 4) ? 16 : 0) + ((v & 3) << 1) + p + (g << 3);
}

// ---------------------------------------------------------------------------
// K0: zero the output accumulator (harness poisons d_out with 0xAA).
__global__ void k_zero(float* __restrict__ out, int n) {
  int i = blockIdx.x * blockDim.x + threadIdx.x;
  if (i < n) out[i] = 0.0f;
}

// ---------------------------------------------------------------------------
// K1: swizzle eW2 (frags 0..63), eb2 (frags 64..65), aW2 (frags 66..67) into
// f16 WMMA B-fragment order in workspace. Fragment f = 512 f16 = 1 KB; the
// consumer lane reads ws + f*512 + lane*16 as one contiguous v16h (32 B).
__global__ void k_swizzle(const float* __restrict__ eW2,
                          const float* __restrict__ eb2,
                          const float* __restrict__ aW2,
                          _Float16* __restrict__ wsf) {
  int tid = blockIdx.x * blockDim.x + threadIdx.x;
  if (tid >= 68 * 512) return;
  int f    = tid >> 9;
  int r    = tid & 511;
  int lane = r >> 4;
  int t    = r & 15;
  int g = lane >> 4, n = lane & 15;
  int v = t >> 1,    p = t & 1;
  int K = a_kidx(t, g);                // B-matrix K index within the 32-block
  float val;
  if (f < 64) {                        // eW2 viewed (1024,32): B[k][o]=eW2[k*32+o]
    int kh = f >> 1, nt = f & 1;
    val = eW2[(kh * 32 + K) * 32 + nt * 16 + n];
  } else if (f < 66) {                 // eb2 reshaped (32,32)
    int nt = f - 64;
    val = eb2[K * 32 + nt * 16 + n];
  } else {                             // aW2 (32,32)
    int nt = f - 66;
    val = aW2[K * 32 + nt * 16 + n];
  }
  wsf[tid] = (_Float16)val;
}

// ---------------------------------------------------------------------------
// K2: edge branch. 256 threads = 8 waves; each wave owns a 16-edge tile and
// produces a 16x32 msg tile with 66 WMMAs (33 K-blocks x 2 N-tiles).
__global__ __launch_bounds__(256) void k_edge(
    const float* __restrict__ x, const int* __restrict__ edge_index,
    const float* __restrict__ edge_attr,
    const float* __restrict__ eW1, const float* __restrict__ eb1,
    const _Float16* __restrict__ wfrag, float* __restrict__ out, int E) {
  __shared__ _Float16 sx[8][16 * 32];  // x[col[e]] rows, f16
  __shared__ _Float16 sh[8][16 * 32];  // per-edge hidden h[e][kh], f16
  __shared__ int      srow[8][16];
  const int w    = threadIdx.x >> 5;
  const int lane = threadIdx.x & 31;   // wave32
  const int g    = lane >> 4;
  const int mn   = lane & 15;          // M row (edge-in-tile) == N col
  const int ebase = (blockIdx.x * 8 + w) * 16;

  // ---- stage: lanes 0-15 -> indices + hidden MLP; lanes 16-31 -> x[col] ----
  {
    int e = ebase + mn;
    if (e >= E) e = E - 1;             // clamp tail; atomics are guarded below
    if (lane < 16) {
      srow[w][mn] = edge_index[e];
      float ea = edge_attr[e];
#pragma unroll
      for (int k = 0; k < CH; ++k) {
        float hv = fmaf(ea, eW1[k], eb1[k]);
        sh[w][mn * 32 + k] = (_Float16)(hv > 0.0f ? hv : 0.0f);
      }
    } else {
      int cc = edge_index[E + e];      // col = edge_index[1][e]
      const float4* xr = (const float4*)(x + cc * CH);
#pragma unroll
      for (int c = 0; c < 8; ++c) {
        float4 xv = xr[c];
        sx[w][mn * 32 + c * 4 + 0] = (_Float16)xv.x;
        sx[w][mn * 32 + c * 4 + 1] = (_Float16)xv.y;
        sx[w][mn * 32 + c * 4 + 2] = (_Float16)xv.z;
        sx[w][mn * 32 + c * 4 + 3] = (_Float16)xv.w;
      }
    }
  }
  __syncthreads();

  // Cached A-fragment of the 16x32 x_col matrix (ISA A lane layout).
  v16h xf;
#pragma unroll
  for (int t = 0; t < 16; ++t) xf[t] = sx[w][mn * 32 + a_kidx(t, g)];

  v8f acc0 = {}, acc1 = {};
  const v16h* wp = (const v16h*)wfrag;           // 32 v16h chunks per fragment
  // 32 hidden K-blocks processed in pairs: two live A fragments (a0, a1) so
  // the WMMA->VALU WAR hazard slots are filled with the other block's work.
  for (int kh = 0; kh < 32; kh += 2) {
    _Float16 h0 = sh[w][mn * 32 + kh];
    _Float16 h1 = sh[w][mn * 32 + kh + 1];
    v16h hv0, hv1;
#pragma unroll
    for (int t = 0; t < 16; ++t) { hv0[t] = h0; hv1[t] = h1; }
    v16h a0 = xf * hv0;                          // v_pk_mul_f16 x8
    v16h a1 = xf * hv1;
    v16h b00 = wp[(kh * 2 + 0) * 32 + lane];
    v16h b01 = wp[(kh * 2 + 1) * 32 + lane];
    v16h b10 = wp[(kh * 2 + 2) * 32 + lane];
    v16h b11 = wp[(kh * 2 + 3) * 32 + lane];
    acc0 = WMMA_F16(a0, b00, acc0);
    acc1 = WMMA_F16(a0, b01, acc1);
    acc0 = WMMA_F16(a1, b10, acc0);
    acc1 = WMMA_F16(a1, b11, acc1);
  }
  // Bias K-block (row block 32 of W = eb2): h == 1, A operand is xf directly.
  {
    v16h b0 = wp[64 * 32 + lane];
    v16h b1 = wp[65 * 32 + lane];
    acc0 = WMMA_F16(xf, b0, acc0);
    acc1 = WMMA_F16(xf, b1, acc1);
  }

  // ---- scatter: C/D layout VGPR r -> M = r + 8g, N = mn ----
#pragma unroll
  for (int r = 0; r < 8; ++r) {
    int me = r + g * 8;
    int e = ebase + me;
    if (e < E) {
      float* dst = out + srow[w][me] * CH;
      atomicAdd(dst + mn,      acc0[r]);
      atomicAdd(dst + 16 + mn, acc1[r]);
    }
  }
}

// ---------------------------------------------------------------------------
// K3: angle branch. feat = relu(ang*aW1+ab1) @ aW2 + ab2 -> scatter to node j.
// One K-block GEMM per 16-angle tile (A = h itself).
__global__ __launch_bounds__(256) void k_angle(
    const int* __restrict__ angle_index, const float* __restrict__ angles,
    const float* __restrict__ aW1, const float* __restrict__ ab1,
    const float* __restrict__ ab2,
    const _Float16* __restrict__ wfrag, float* __restrict__ out, int A) {
  __shared__ _Float16 sh[8][16 * 32];
  __shared__ int      sj[8][16];
  const int w    = threadIdx.x >> 5;
  const int lane = threadIdx.x & 31;
  const int g    = lane >> 4;
  const int mn   = lane & 15;
  const int abase = (blockIdx.x * 8 + w) * 16;

  if (lane < 16) {
    int e = abase + mn;
    if (e >= A) e = A - 1;
    sj[w][mn] = angle_index[A + e];    // j = angle_index[1][e]
    float an = angles[e];
#pragma unroll
    for (int k = 0; k < CH; ++k) {
      float hv = fmaf(an, aW1[k], ab1[k]);
      sh[w][mn * 32 + k] = (_Float16)(hv > 0.0f ? hv : 0.0f);
    }
  }
  __syncthreads();

  v16h a;
#pragma unroll
  for (int t = 0; t < 16; ++t) a[t] = sh[w][mn * 32 + a_kidx(t, g)];

  const v16h* wp = (const v16h*)wfrag;
  v16h b0 = wp[66 * 32 + lane];
  v16h b1 = wp[67 * 32 + lane];
  v8f acc0 = {}, acc1 = {};
  acc0 = WMMA_F16(a, b0, acc0);
  acc1 = WMMA_F16(a, b1, acc1);

  float bias0 = ab2[mn], bias1 = ab2[16 + mn];
#pragma unroll
  for (int r = 0; r < 8; ++r) {
    int me = r + g * 8;
    int e = abase + me;
    if (e < A) {
      float* dst = out + sj[w][me] * CH;
      atomicAdd(dst + mn,      acc0[r] + bias0);
      atomicAdd(dst + 16 + mn, acc1[r] + bias1);
    }
  }
}

// ---------------------------------------------------------------------------
extern "C" void kernel_launch(void* const* d_in, const int* in_sizes, int n_in,
                              void* d_out, int out_size, void* d_ws,
                              size_t ws_size, hipStream_t stream) {
  const float* x           = (const float*)d_in[0];
  const int*   edge_index  = (const int*)d_in[1];
  const float* edge_attr   = (const float*)d_in[2];
  const int*   angle_index = (const int*)d_in[3];
  const float* angles      = (const float*)d_in[4];
  const float* eW1 = (const float*)d_in[5];
  const float* eb1 = (const float*)d_in[6];
  const float* eW2 = (const float*)d_in[7];
  const float* eb2 = (const float*)d_in[8];
  const float* aW1 = (const float*)d_in[9];
  const float* ab1 = (const float*)d_in[10];
  const float* aW2 = (const float*)d_in[11];
  const float* ab2 = (const float*)d_in[12];
  float*     out = (float*)d_out;
  _Float16*  wsf = (_Float16*)d_ws;    // needs 68*512*2 = 69632 bytes

  const int E = in_sizes[1] / 2;
  const int A = in_sizes[3] / 3;

  k_zero<<<(out_size + 255) / 256, 256, 0, stream>>>(out, out_size);
  k_swizzle<<<(68 * 512 + 255) / 256, 256, 0, stream>>>(eW2, eb2, aW2, wsf);
  k_edge<<<(E + 127) / 128, 256, 0, stream>>>(x, edge_index, edge_attr, eW1,
                                              eb1, wsf, out, E);
  k_angle<<<(A + 127) / 128, 256, 0, stream>>>(angle_index, angles, aW1, ab1,
                                               ab2, wsf, out, A);
}
